// crossFrequencyAttention_73005854098213
// MI455X (gfx1250) — matrix operand order repaired and checked by reading, not compiled, not run on hardware
//
#include <hip/hip_runtime.h>
#include <hip/hip_bf16.h>

#define B_   8
#define L_   2048
#define D_   512
#define F_   1025
#define FP_  1056            // padded row length / K dim: 66*16, divisible by 32
#define MP_  1152            // padded freq row count: 9*128 (exact 128-row tiles)
#define M_   (B_*L_)         // 16384 rows for projection GEMMs

typedef __attribute__((ext_vector_type(16))) _Float16 v16h;
typedef __attribute__((ext_vector_type(8)))  _Float16 h8;
typedef __attribute__((ext_vector_type(4)))  _Float16 h4;
typedef __attribute__((ext_vector_type(8)))  float    v8f;

#define LDSTRIDE 40   // f16 row stride in LDS tiles: 80B -> conflict-free b128 frag reads

// ---------------------------------------------------------------------------
// WMMA wrapper: D = A(16x32 f16) * B(32x16 f16) + C(16x16 f32)
// ---------------------------------------------------------------------------
__device__ __forceinline__ v8f wmma16(v16h a, v16h b, v8f c) {
  return __builtin_amdgcn_wmma_f32_16x16x32_f16(false, a, false, b,
                                                (short)0, c, false, false);
}

// ---------------------------------------------------------------------------
// Fragment reads from LDS tiles (layouts per CDNA5 ISA 7.12.2).
// A tile in LDS: [rows][LDSTRIDE] f16 row-major (K contiguous).
// B tile in LDS: TRANSPOSED [N][LDSTRIDE] f16 (K contiguous per output col).
// Each fragment = 2x ds_load_b128 per lane, conflict-free with 80B row pitch.
// ---------------------------------------------------------------------------
__device__ __forceinline__ v16h lds_read_a(const _Float16* base, int rowtile) {
  int lane = threadIdx.x & 31;
  const _Float16* p = base + (rowtile * 16 + (lane & 15)) * LDSTRIDE
                           + ((lane & 16) ? 8 : 0);
  h8 lo = *(const h8*)(p);
  h8 hi = *(const h8*)(p + 16);
  return __builtin_shufflevector(lo, hi, 0,1,2,3,4,5,6,7,8,9,10,11,12,13,14,15);
}

__device__ __forceinline__ v16h lds_read_b(const _Float16* base, int ntile) {
  int lane = threadIdx.x & 31;
  const _Float16* p = base + (ntile * 16 + (lane & 15)) * LDSTRIDE
                           + ((lane & 16) ? 8 : 0);
  h8 lo = *(const h8*)(p);
  h8 hi = *(const h8*)(p + 16);
  return __builtin_shufflevector(lo, hi, 0,1,2,3,4,5,6,7,8,9,10,11,12,13,14,15);
}

// ---------------------------------------------------------------------------
// Kernel 1/7: C[M,N] = A[M,K] @ W[K,N] + bias   (f32 in/out, f16 WMMA inside)
// block = 256 (8 waves), block tile 128x64, K-step 32.
// Double-buffered LDS staging; stage = load-all-then-store to avoid per-load
// s_wait_loadcnt serialization.
// grid = (M/128, N/64)
// ---------------------------------------------------------------------------
__device__ __forceinline__ void stage_real(const float* __restrict__ A,
                                           const float* __restrict__ W,
                                           _Float16* As, _Float16* Bs,
                                           int row0, int col0, int K, int N,
                                           int k0, int tid) {
  float4 va[4], vb[2];
  // Load phase: all global b128 loads into distinct registers
#pragma unroll
  for (int i = 0; i < 4; ++i) {
    int idx = tid + i * 256;
    va[i] = *(const float4*)(A + (size_t)(row0 + (idx >> 3)) * K + k0 + (idx & 7) * 4);
  }
#pragma unroll
  for (int i = 0; i < 2; ++i) {
    int idx = tid + i * 256;
    vb[i] = *(const float4*)(W + (size_t)(k0 + (idx >> 4)) * N + col0 + (idx & 15) * 4);
  }
  // Store phase: convert f32->f16 and write LDS tiles
#pragma unroll
  for (int i = 0; i < 4; ++i) {
    int idx = tid + i * 256;
    int r   = idx >> 3;
    int kq  = (idx & 7) * 4;
    *(h4*)(&As[r * LDSTRIDE + kq]) =
        (h4){(_Float16)va[i].x, (_Float16)va[i].y, (_Float16)va[i].z, (_Float16)va[i].w};
  }
#pragma unroll
  for (int i = 0; i < 2; ++i) {
    int idx = tid + i * 256;
    int kr  = idx >> 4;
    int nq  = (idx & 15) * 4;
    Bs[(nq + 0) * LDSTRIDE + kr] = (_Float16)vb[i].x;
    Bs[(nq + 1) * LDSTRIDE + kr] = (_Float16)vb[i].y;
    Bs[(nq + 2) * LDSTRIDE + kr] = (_Float16)vb[i].z;
    Bs[(nq + 3) * LDSTRIDE + kr] = (_Float16)vb[i].w;
  }
}

__global__ void __launch_bounds__(256)
gemm_bias_kernel(const float* __restrict__ A, const float* __restrict__ W,
                 const float* __restrict__ bias, float* __restrict__ out,
                 int M, int K, int N) {
  __shared__ __align__(16) _Float16 As[2][128 * LDSTRIDE];
  __shared__ __align__(16) _Float16 Bs[2][64 * LDSTRIDE];
  const int tid  = threadIdx.x;
  const int wave = tid >> 5;
  const int lane = tid & 31;
  const int row0 = blockIdx.x * 128;
  const int col0 = blockIdx.y * 64;
  const int nsteps = K >> 5;

  stage_real(A, W, As[0], Bs[0], row0, col0, K, N, 0, tid);
  __syncthreads();

  v8f acc[4] = {};
  for (int s = 0; s < nsteps; ++s) {
    int cur = s & 1, nxt = cur ^ 1;
    if (s + 1 < nsteps) {
      __builtin_prefetch(A + (size_t)(row0 + (tid >> 3)) * K + (s + 2) * 32, 0, 1);
      stage_real(A, W, As[nxt], Bs[nxt], row0, col0, K, N, (s + 1) * 32, tid);
    }
    v16h a = lds_read_a(As[cur], wave);
    v16h b[4];
#pragma unroll
    for (int j = 0; j < 4; ++j) b[j] = lds_read_b(Bs[cur], j);
#pragma unroll
    for (int j = 0; j < 4; ++j) acc[j] = wmma16(a, b[j], acc[j]);
    __syncthreads();
  }
  int n = lane & 15, mb = (lane & 16) ? 8 : 0;
#pragma unroll
  for (int j = 0; j < 4; ++j)
#pragma unroll
    for (int v = 0; v < 8; ++v) {
      int c = col0 + j * 16 + n;
      out[(size_t)(row0 + wave * 16 + mb + v) * N + c] = acc[j][v] + bias[c];
    }
}

// ---------------------------------------------------------------------------
// In-LDS 2048-point radix-2 FFT stages. dir = -1 forward, +1 inverse.
// ---------------------------------------------------------------------------
__device__ __forceinline__ void fft2048_stages(float2* lds, float dir) {
#pragma unroll 1
  for (int s = 0; s < 11; ++s) {
    int half = 1 << s;
    float angstep = dir * 3.14159265358979323f / (float)half;
    for (int t = threadIdx.x; t < 1024; t += 256) {
      int kk = t & (half - 1);
      int i0 = ((t >> s) << (s + 1)) | kk;
      float ang = angstep * (float)kk;
      float wi, wr;
      __sincosf(ang, &wi, &wr);
      float2 u = lds[i0];
      float2 v = lds[i0 + half];
      float vr = v.x * wr - v.y * wi;
      float vi = v.x * wi + v.y * wr;
      lds[i0]        = make_float2(u.x + vr, u.y + vi);
      lds[i0 + half] = make_float2(u.x - vr, u.y - vi);
    }
    __syncthreads();
  }
}

// ---------------------------------------------------------------------------
// Kernel 2: rfft of one (b,d) column of sig[B][L][D].
// kmajor==0: out[b][f][D_], f < rows (zero-padded f >= F_)
// kmajor==1: out[b][d][FP_] (B-operand layout, zero-padded g >= F_)
// grid = B_*D_, block = 256
// ---------------------------------------------------------------------------
__global__ void __launch_bounds__(256)
rfft_kernel(const float* __restrict__ sig, float2* __restrict__ out,
            int kmajor, int rows) {
  __shared__ float2 lds[2048];
  int b = blockIdx.x >> 9;
  int d = blockIdx.x & 511;
  const float* col = sig + (size_t)b * (L_ * D_) + d;
  for (int i = threadIdx.x; i < 2048; i += 256) {
    int j = __brev((unsigned)i) >> 21;
    lds[j] = make_float2(col[(size_t)i * D_], 0.0f);
  }
  __syncthreads();
  fft2048_stages(lds, -1.0f);
  if (kmajor == 0) {
    float2* o = out + (size_t)b * ((size_t)rows * D_) + d;
    for (int f = threadIdx.x; f < rows; f += 256)
      o[(size_t)f * D_] = (f < F_) ? lds[f] : make_float2(0.f, 0.f);
  } else {
    float2* o = out + ((size_t)b * D_ + d) * FP_;
    for (int f = threadIdx.x; f < FP_; f += 256)
      o[f] = (f < F_) ? lds[f] : make_float2(0.f, 0.f);
  }
}

// ---------------------------------------------------------------------------
// Kernel 3: att[b][f][g] = scale * | sum_d Qf[b,f,d] * Kf[b,g,d] |
// Complex GEMM, double-buffered LDS staging. Block tile 128(f) x 32(g).
// 4 real accumulators per complex tile (f16 WMMA has no operand-NEG).
// grid = (MP_/128, FP_/32, B_)
// ---------------------------------------------------------------------------
__device__ __forceinline__ void stage_cx_scores(const float2* __restrict__ Qb,
                                                const float2* __restrict__ Kb,
                                                _Float16* Ar, _Float16* Ai,
                                                _Float16* Br, _Float16* Bi,
                                                int f0, int g0, int k0, int tid) {
  float4 qa[8], kb[2];
  // Load phase: A = 128x32 complex (2 float4 per thread-iter), B = 32x32 complex
#pragma unroll
  for (int i = 0; i < 4; ++i) {
    int idx = tid + i * 256;
    const float4* p = (const float4*)(Qb + (size_t)(f0 + (idx >> 3)) * D_
                                         + k0 + (idx & 7) * 4);
    qa[2 * i]     = p[0];
    qa[2 * i + 1] = p[1];
  }
  {
    const float4* p = (const float4*)(Kb + (size_t)(k0 + (tid >> 3)) * FP_
                                         + g0 + (tid & 7) * 4);
    kb[0] = p[0];
    kb[1] = p[1];
  }
  // Store phase
#pragma unroll
  for (int i = 0; i < 4; ++i) {
    int idx = tid + i * 256;
    int r   = idx >> 3;
    int dq  = (idx & 7) * 4;
    float4 u = qa[2 * i], v = qa[2 * i + 1];
    *(h4*)(&Ar[r * LDSTRIDE + dq]) =
        (h4){(_Float16)u.x, (_Float16)u.z, (_Float16)v.x, (_Float16)v.z};
    *(h4*)(&Ai[r * LDSTRIDE + dq]) =
        (h4){(_Float16)u.y, (_Float16)u.w, (_Float16)v.y, (_Float16)v.w};
  }
  {
    int kr = tid >> 3;
    int gq = (tid & 7) * 4;
    float4 u = kb[0], v = kb[1];
    Br[(gq + 0) * LDSTRIDE + kr] = (_Float16)u.x;
    Bi[(gq + 0) * LDSTRIDE + kr] = (_Float16)u.y;
    Br[(gq + 1) * LDSTRIDE + kr] = (_Float16)u.z;
    Bi[(gq + 1) * LDSTRIDE + kr] = (_Float16)u.w;
    Br[(gq + 2) * LDSTRIDE + kr] = (_Float16)v.x;
    Bi[(gq + 2) * LDSTRIDE + kr] = (_Float16)v.y;
    Br[(gq + 3) * LDSTRIDE + kr] = (_Float16)v.z;
    Bi[(gq + 3) * LDSTRIDE + kr] = (_Float16)v.w;
  }
}

__global__ void __launch_bounds__(256)
scores_kernel(const float2* __restrict__ Qf, const float2* __restrict__ Kf,
              float* __restrict__ att) {
  __shared__ __align__(16) _Float16 Ar[2][128 * LDSTRIDE];
  __shared__ __align__(16) _Float16 Ai[2][128 * LDSTRIDE];
  __shared__ __align__(16) _Float16 Br[2][32 * LDSTRIDE];
  __shared__ __align__(16) _Float16 Bi[2][32 * LDSTRIDE];
  const int tid  = threadIdx.x;
  const int wave = tid >> 5;
  const int lane = tid & 31;
  const int b  = blockIdx.z;
  const int f0 = blockIdx.x * 128;
  const int g0 = blockIdx.y * 32;
  const float2* Qb = Qf + (size_t)b * MP_ * D_;
  const float2* Kb = Kf + (size_t)b * D_ * FP_;
  const int nsteps = D_ >> 5;   // 16

  stage_cx_scores(Qb, Kb, Ar[0], Ai[0], Br[0], Bi[0], f0, g0, 0, tid);
  __syncthreads();

  v8f rr[2] = {}, ii[2] = {}, ri[2] = {}, ir[2] = {};
  for (int s = 0; s < nsteps; ++s) {
    int cur = s & 1, nxt = cur ^ 1;
    if (s + 1 < nsteps)
      stage_cx_scores(Qb, Kb, Ar[nxt], Ai[nxt], Br[nxt], Bi[nxt],
                      f0, g0, (s + 1) * 32, tid);
    v16h ar = lds_read_a(Ar[cur], wave);
    v16h ai = lds_read_a(Ai[cur], wave);
    v16h br[2], bi[2];
#pragma unroll
    for (int j = 0; j < 2; ++j) {
      br[j] = lds_read_b(Br[cur], j);
      bi[j] = lds_read_b(Bi[cur], j);
    }
#pragma unroll
    for (int j = 0; j < 2; ++j) {
      rr[j] = wmma16(ar, br[j], rr[j]);
      ii[j] = wmma16(ai, bi[j], ii[j]);
      ri[j] = wmma16(ar, bi[j], ri[j]);
      ir[j] = wmma16(ai, br[j], ir[j]);
    }
    __syncthreads();
  }
  const float scale = 0.04419417382415922f;   // 512^-0.5
  int n = lane & 15, mb = (lane & 16) ? 8 : 0;
  float* ob = att + (size_t)b * MP_ * FP_;
#pragma unroll
  for (int j = 0; j < 2; ++j)
#pragma unroll
    for (int v = 0; v < 8; ++v) {
      float sr = rr[j][v] - ii[j][v];
      float si = ri[j][v] + ir[j][v];
      ob[(size_t)(f0 + wave * 16 + mb + v) * FP_ + g0 + j * 16 + n] =
          scale * sqrtf(sr * sr + si * si);
    }
}

// ---------------------------------------------------------------------------
// Kernel 4: row softmax over g<F_, emit f16 attn with zero padding g>=F_.
// grid = (MP_, B_), block = 256
// ---------------------------------------------------------------------------
__global__ void __launch_bounds__(256)
softmax_kernel(const float* __restrict__ att, _Float16* __restrict__ attn) {
  __shared__ float red[256];
  int b = blockIdx.y, f = blockIdx.x;
  const float* row = att + ((size_t)b * MP_ + f) * FP_;
  _Float16* orow   = attn + ((size_t)b * MP_ + f) * FP_;

  float m = -1e30f;
  for (int g = threadIdx.x; g < F_; g += 256) m = fmaxf(m, row[g]);
  red[threadIdx.x] = m; __syncthreads();
  for (int s = 128; s > 0; s >>= 1) {
    if (threadIdx.x < s) red[threadIdx.x] = fmaxf(red[threadIdx.x], red[threadIdx.x + s]);
    __syncthreads();
  }
  m = red[0]; __syncthreads();

  float sum = 0.f;
  for (int g = threadIdx.x; g < F_; g += 256) sum += __expf(row[g] - m);
  red[threadIdx.x] = sum; __syncthreads();
  for (int s = 128; s > 0; s >>= 1) {
    if (threadIdx.x < s) red[threadIdx.x] += red[threadIdx.x + s];
    __syncthreads();
  }
  float inv = 1.0f / red[0];

  for (int g = threadIdx.x; g < FP_; g += 256)
    orow[g] = (g < F_) ? (_Float16)(__expf(row[g] - m) * inv) : (_Float16)0.f;
}

// ---------------------------------------------------------------------------
// Kernel 5: ctx[b][f][d] = sum_g attn[b,f,g] * Vf[b,g,d]  (real f16 x complex)
// Double-buffered LDS staging. Block tile 128(f) x 64(d), K-step 32 over g.
// grid = (MP_/128, D_/64, B_)
// ---------------------------------------------------------------------------
__device__ __forceinline__ void stage_ctx(const _Float16* __restrict__ Ab,
                                          const float2* __restrict__ Vb,
                                          _Float16* As, _Float16* Br, _Float16* Bi,
                                          int f0, int d0, int k0, int tid) {
  h8 ha[2];
  float4 vv[4];
  // Load phase
#pragma unroll
  for (int i = 0; i < 2; ++i) {
    int idx = tid + i * 256;
    ha[i] = *(const h8*)(Ab + (size_t)(f0 + (idx >> 2)) * FP_ + k0 + (idx & 3) * 8);
  }
#pragma unroll
  for (int i = 0; i < 2; ++i) {
    int idx = tid + i * 256;
    const float4* p = (const float4*)(Vb + (size_t)(k0 + (idx >> 4)) * D_
                                         + d0 + (idx & 15) * 4);
    vv[2 * i]     = p[0];
    vv[2 * i + 1] = p[1];
  }
  // Store phase
#pragma unroll
  for (int i = 0; i < 2; ++i) {
    int idx = tid + i * 256;
    *(h8*)(&As[(idx >> 2) * LDSTRIDE + (idx & 3) * 8]) = ha[i];
  }
#pragma unroll
  for (int i = 0; i < 2; ++i) {
    int idx = tid + i * 256;
    int kr  = idx >> 4;
    int dq  = (idx & 15) * 4;
    float4 u = vv[2 * i], v = vv[2 * i + 1];
    Br[(dq + 0) * LDSTRIDE + kr] = (_Float16)u.x;
    Bi[(dq + 0) * LDSTRIDE + kr] = (_Float16)u.y;
    Br[(dq + 1) * LDSTRIDE + kr] = (_Float16)u.z;
    Bi[(dq + 1) * LDSTRIDE + kr] = (_Float16)u.w;
    Br[(dq + 2) * LDSTRIDE + kr] = (_Float16)v.x;
    Bi[(dq + 2) * LDSTRIDE + kr] = (_Float16)v.y;
    Br[(dq + 3) * LDSTRIDE + kr] = (_Float16)v.z;
    Bi[(dq + 3) * LDSTRIDE + kr] = (_Float16)v.w;
  }
}

__global__ void __launch_bounds__(256)
context_kernel(const _Float16* __restrict__ attn, const float2* __restrict__ Vf,
               float2* __restrict__ ctx) {
  __shared__ __align__(16) _Float16 As[2][128 * LDSTRIDE];
  __shared__ __align__(16) _Float16 Br[2][64 * LDSTRIDE];
  __shared__ __align__(16) _Float16 Bi[2][64 * LDSTRIDE];
  const int tid  = threadIdx.x;
  const int wave = tid >> 5;
  const int lane = tid & 31;
  const int b  = blockIdx.z;
  const int f0 = blockIdx.x * 128;
  const int d0 = blockIdx.y * 64;
  const _Float16* Ab = attn + (size_t)b * MP_ * FP_;
  const float2*   Vb = Vf + (size_t)b * FP_ * D_;
  const int nsteps = FP_ >> 5;   // 33

  stage_ctx(Ab, Vb, As[0], Br[0], Bi[0], f0, d0, 0, tid);
  __syncthreads();

  v8f accr[4] = {}, acci[4] = {};
  for (int s = 0; s < nsteps; ++s) {
    int cur = s & 1, nxt = cur ^ 1;
    if (s + 1 < nsteps)
      stage_ctx(Ab, Vb, As[nxt], Br[nxt], Bi[nxt], f0, d0, (s + 1) * 32, tid);
    v16h a = lds_read_a(As[cur], wave);
    v16h br[4], bi[4];
#pragma unroll
    for (int j = 0; j < 4; ++j) {
      br[j] = lds_read_b(Br[cur], j);
      bi[j] = lds_read_b(Bi[cur], j);
    }
#pragma unroll
    for (int j = 0; j < 4; ++j) {
      accr[j] = wmma16(a, br[j], accr[j]);
      acci[j] = wmma16(a, bi[j], acci[j]);
    }
    __syncthreads();
  }
  int n = lane & 15, mb = (lane & 16) ? 8 : 0;
  float2* obase = ctx + (size_t)b * MP_ * D_;
#pragma unroll
  for (int j = 0; j < 4; ++j)
#pragma unroll
    for (int v = 0; v < 8; ++v)
      obase[(size_t)(f0 + wave * 16 + mb + v) * D_ + d0 + j * 16 + n] =
          make_float2(accr[j][v], acci[j][v]);
}

// ---------------------------------------------------------------------------
// Kernel 6: irfft(n=2048) of one (b,d) column of ctx[B][MP_][D] -> out[B][L][D]
// grid = B_*D_, block = 256
// ---------------------------------------------------------------------------
__global__ void __launch_bounds__(256)
irfft_kernel(const float2* __restrict__ ctx, float* __restrict__ out) {
  __shared__ float2 lds[2048];
  int b = blockIdx.x >> 9;
  int d = blockIdx.x & 511;
  const float2* col = ctx + (size_t)b * (MP_ * D_) + d;
  for (int k = threadIdx.x; k < 2048; k += 256) {
    float2 v;
    if (k <= 1024) {
      v = col[(size_t)k * D_];
    } else {
      float2 t = col[(size_t)(2048 - k) * D_];
      v = make_float2(t.x, -t.y);                // conj symmetry
    }
    lds[__brev((unsigned)k) >> 21] = v;
  }
  __syncthreads();
  fft2048_stages(lds, +1.0f);
  float* o = out + (size_t)b * (L_ * D_) + d;
  const float invN = 1.0f / 2048.0f;
  for (int l = threadIdx.x; l < 2048; l += 256)
    o[(size_t)l * D_] = lds[l].x * invN;
}

// ---------------------------------------------------------------------------
// Orchestration. 4 workspace buffers of SZ bytes, time-multiplexed:
//   buf0: Qt/Kt/Vt time-domain -> att_abs -> ctxT
//   buf1: Qf [B][MP_][D] cx    -> attn f16
//   buf2: Kf [B][D][FP_] cx    -> ctx [B][MP_][D] cx
//   buf3: Vf [B][FP_][D] cx
// Peak ws use: 4*SZ = 155.7 MB (SZ = B*MP_*FP_*4 = 38,928,384, 256B-aligned).
// ---------------------------------------------------------------------------
extern "C" void kernel_launch(void* const* d_in, const int* in_sizes, int n_in,
                              void* d_out, int out_size, void* d_ws, size_t ws_size,
                              hipStream_t stream) {
  (void)in_sizes; (void)n_in; (void)out_size; (void)ws_size;
  const float* x  = (const float*)d_in[0];
  const float* Wq = (const float*)d_in[1];
  const float* bq = (const float*)d_in[2];
  const float* Wk = (const float*)d_in[3];
  const float* bk = (const float*)d_in[4];
  const float* Wv = (const float*)d_in[5];
  const float* bv = (const float*)d_in[6];
  const float* Wo = (const float*)d_in[7];
  const float* bo = (const float*)d_in[8];
  float* out = (float*)d_out;

  const size_t SZ = (size_t)B_ * MP_ * FP_ * sizeof(float);
  char* w = (char*)d_ws;
  float*  buf0 = (float*)(w);
  float2* Qf   = (float2*)(w + SZ);
  float2* Kf   = (float2*)(w + 2 * SZ);
  float2* Vf   = (float2*)(w + 3 * SZ);

  dim3 gGemm(M_ / 128, D_ / 64);            // (128, 8)
  dim3 gFft(B_ * D_);                       // 4096
  dim3 gScore(MP_ / 128, FP_ / 32, B_);     // (9, 33, 8)
  dim3 gSoft(MP_, B_);                      // (1152, 8)
  dim3 gCtx(MP_ / 128, D_ / 64, B_);        // (9, 8, 8)

  // Q path (freq rows padded to MP_ for exact 128-row GEMM tiles)
  gemm_bias_kernel<<<gGemm, 256, 0, stream>>>(x, Wq, bq, buf0, M_, D_, D_);
  rfft_kernel<<<gFft, 256, 0, stream>>>(buf0, Qf, 0, MP_);
  // K path (d-major freq layout = WMMA B operand)
  gemm_bias_kernel<<<gGemm, 256, 0, stream>>>(x, Wk, bk, buf0, M_, D_, D_);
  rfft_kernel<<<gFft, 256, 0, stream>>>(buf0, Kf, 1, FP_);
  // V path
  gemm_bias_kernel<<<gGemm, 256, 0, stream>>>(x, Wv, bv, buf0, M_, D_, D_);
  rfft_kernel<<<gFft, 256, 0, stream>>>(buf0, Vf, 0, FP_);

  // complex attention scores -> |.|*scale
  scores_kernel<<<gScore, 256, 0, stream>>>(Qf, Kf, buf0);
  // softmax rows -> f16 attn (reuses Qf space)
  softmax_kernel<<<gSoft, 256, 0, stream>>>(buf0, (_Float16*)Qf);
  // context = attn @ Vf (reuses Kf space)
  context_kernel<<<gCtx, 256, 0, stream>>>((const _Float16*)Qf, Vf, Kf);
  // back to time domain (reuses buf0)
  irfft_kernel<<<gFft, 256, 0, stream>>>(Kf, buf0);
  // output projection
  gemm_bias_kernel<<<gGemm, 256, 0, stream>>>(buf0, Wo, bo, out, M_, D_, D_);
}
